// PoissonEMModule_65824668779086
// MI455X (gfx1250) — matrix operand order repaired and checked by reading, not compiled, not run on hardware
//
#include <hip/hip_runtime.h>
#include <hip/hip_bf16.h>

// ---------------- problem geometry (from reference) ----------------
#define N0      16
#define N1      200
#define N2      200
#define NPHI    190
#define NRAD    223
#define NT      351
#define SLICE   (N1 * N2)            // 40000
#define NVOX    (N0 * SLICE)         // 640000
#define SINO_Z  (NPHI * NRAD)        // 42370
#define VOXF    2.0f
#define STEPF   2.0f
#define ORGF    (-199.0f)            // (-N1/2 + 0.5) * VOX
#define DRF     (400.0f / 222.0f)    // radial spacing
#define RADIUS  4
#define KW      9
#define SIGMAF  (4.5f / (2.35f * 2.0f))
#define PI_F    3.14159265358979323846f

#define NTILES  (SLICE / 16)         // 2500 column tiles for the z-filter
#define TPW     4                    // tiles per wave (pipelined)

typedef _Float16     half16  __attribute__((ext_vector_type(16)));
typedef float        float8  __attribute__((ext_vector_type(8)));
typedef unsigned int uint32x4 __attribute__((ext_vector_type(4)));
typedef int          int32x4  __attribute__((ext_vector_type(4)));
typedef int          int32x8  __attribute__((ext_vector_type(8)));

__device__ __forceinline__ int reflect(int p, int L) {
  p = (p < 0) ? (-1 - p) : p;
  p = (p > L - 1) ? (2 * L - 1 - p) : p;
  return p;
}

__device__ __forceinline__ int iclamp(int v, int lo, int hi) {
  return (v < lo) ? lo : ((v > hi) ? hi : v);
}

__device__ __forceinline__ void make_gw(float* g) {
  float s = 0.f;
  #pragma unroll
  for (int k = 0; k < KW; ++k) {
    float u = (float)(k - RADIUS) / SIGMAF;
    g[k] = __expf(-0.5f * u * u);
    s += g[k];
  }
  float inv = 1.f / s;
  #pragma unroll
  for (int k = 0; k < KW; ++k) g[k] *= inv;
}

// full-correlation value c[p] with zero padding (adjoint-smoothing core).
// Branchless: clamp the address (always loadable), zero the weight when OOB,
// so the 9 loads issue as one unconditional clause instead of 9 exec-masked
// regions.
__device__ __forceinline__ float conv_full(const float* row, int stride, int p,
                                           const float* g, int L) {
  float a = 0.f;
  #pragma unroll
  for (int k = 0; k < KW; ++k) {
    int src  = p - (k - RADIUS);
    int srcc = iclamp(src, 0, L - 1);
    float w  = (src == srcc) ? g[k] : 0.f;
    a += w * row[srcc * stride];
  }
  return a;
}

// adjoint of symmetric-pad 9-tap smoothing along one axis:
// z[j] = c[j] + (j<=3 ? c[-1-j] : 0) + (j>=L-4 ? c[2L-1-j] : 0)
__device__ __forceinline__ float adj_axis(const float* row, int stride, int i,
                                          const float* g, int L) {
  float v = conv_full(row, stride, i, g, L);
  if (i <= RADIUS - 1) v += conv_full(row, stride, -1 - i, g, L);
  if (i >= L - RADIUS) v += conv_full(row, stride, 2 * L - 1 - i, g, L);
  return v;
}

// ---------------- axis-2 / axis-1 forward smoothing ----------------
__global__ __launch_bounds__(256) void k_smooth2(const float* __restrict__ in,
                                                 float* __restrict__ out) {
  int idx = blockIdx.x * blockDim.x + threadIdx.x;
  if (idx >= NVOX) return;
  float g[KW]; make_gw(g);
  int j = idx % N2;
  const float* row = in + (idx - j);
  float acc = 0.f;
  #pragma unroll
  for (int k = 0; k < KW; ++k) acc += g[k] * row[reflect(j + k - RADIUS, N2)];
  out[idx] = acc;
}

__global__ __launch_bounds__(256) void k_smooth1(const float* __restrict__ in,
                                                 float* __restrict__ out) {
  int idx = blockIdx.x * blockDim.x + threadIdx.x;
  if (idx >= NVOX) return;
  float g[KW]; make_gw(g);
  int i = (idx / N2) % N1;
  const float* col = in + (idx - i * N2);
  float acc = 0.f;
  #pragma unroll
  for (int k = 0; k < KW; ++k) acc += g[k] * col[reflect(i + k - RADIUS, N1) * N2];
  out[idx] = acc;
}

// ---------------- TDM tile load: 16 z-rows x 16 cols (f32) -> LDS ----------------
__device__ __forceinline__ void tdm_issue_tile(const float* src_tile_base,
                                               unsigned lds_byte_off) {
  unsigned long long gaddr = (unsigned long long)src_tile_base;
  uint32x4 d0;
  d0[0] = 1u;                                            // count=1 valid descriptor
  d0[1] = lds_byte_off;                                  // lds_addr (bytes)
  d0[2] = (unsigned)(gaddr & 0xFFFFFFFFull);             // global_addr[31:0]
  d0[3] = (unsigned)((gaddr >> 32) & 0x1FFFFFFull) | (2u << 30);  // [56:32] | type=2
  int32x8 d1;
  d1[0] = (int)(2u << 16);                               // data_size = 4B
  d1[1] = (int)(0x9C40u << 16);                          // tensor_dim0 = 40000 (lo16)
  d1[2] = (int)(16u << 16);                              // dim0 hi=0 | tensor_dim1=16
  d1[3] = (int)(16u << 16);                              // dim1 hi=0 | tile_dim0=16
  d1[4] = 16;                                            // tile_dim1=16 | tile_dim2=0
  d1[5] = SLICE;                                         // tensor_dim0_stride = 40000
  d1[6] = 0;
  d1[7] = 0;
  int32x4 d2 = {0, 0, 0, 0}, d3 = {0, 0, 0, 0};
#if __has_include(<hip/amd_detail/amd_gfx1250_TDM.h>)
  int32x8 dz = {0, 0, 0, 0, 0, 0, 0, 0};
  __builtin_amdgcn_tensor_load_to_lds(d0, d1, d2, d3, dz, 0);   // clang-23 (6-arg)
#else
  __builtin_amdgcn_tensor_load_to_lds(d0, d1, d2, d3, 0);       // ROCm 7.2 (5-arg)
#endif
}

// ---------------- axis-0 (z, length 16) smoothing via WMMA + pipelined TDM ----------
// out[16 x 40000] = G (16x16 banded, symmetric-pad Gaussian) * in[16 x 40000].
// Each wave owns TPW consecutive 16-column tiles and double-buffers the TDM
// loads in LDS: issue load(i+1) before waiting (s_wait_tensorcnt 1 — TDM ops
// complete in order per wave), drain to 0 only on the last tile.
__global__ __launch_bounds__(256) void k_zfilt_wmma(const float* __restrict__ in,
                                                    float* __restrict__ out,
                                                    int transpose) {
  __shared__ float tdm_buf[8 * 2 * 256];       // 8 waves * 2 buffers * 16x16 f32 = 16 KB
  float g[KW]; make_gw(g);

  const int lane = threadIdx.x & 31;
  const int wave = threadIdx.x >> 5;
  const int m    = lane & 15;                  // M (A) / N (B,D) index
  const int grp  = lane >> 4;                  // half-wave group

  // A fragment (16-bit A 16x32 layout): half h of lane -> K index:
  //   h<8 : K = h + 8*grp ;  h>=8 : K = h + 8 + 8*grp  (K>=16 -> zero pad)
  half16 a;
  #pragma unroll
  for (int h = 0; h < 16; ++h) {
    int K = (h < 8) ? (h + 8 * grp) : (h + 8 + 8 * grp);
    float v = 0.f;
    if (K < N0) {
      int i = transpose ? K : m;               // A = G0 or G0^T
      int j = transpose ? m : K;
      #pragma unroll
      for (int k = 0; k < KW; ++k)
        if (reflect(i + k - RADIUS, N0) == j) v += g[k];
    }
    a[h] = (_Float16)v;
  }

  const int wid   = blockIdx.x * (blockDim.x >> 5) + wave;  // global wave id
  const int tbase = wid * TPW;
  if (tbase >= NTILES) return;                 // whole-wave uniform

  const unsigned lds0 = (unsigned)(size_t)(&tdm_buf[0]) + (unsigned)(wave * 2048);
  const float* lbuf0 = tdm_buf + wave * 512;

  // prologue: stage tile 0 of this wave's strip
  tdm_issue_tile(in + tbase * 16, lds0);

  #pragma unroll
  for (int it = 0; it < TPW; ++it) {
    const int tile = tbase + it;
    if (tile >= NTILES) break;                 // uniform per wave
    const bool more = (it + 1 < TPW) && (tile + 1 < NTILES);
    if (more) {
      // issue next tile into the other buffer, then wait for the current one
      tdm_issue_tile(in + (tile + 1) * 16, lds0 + (((it + 1) & 1) ? 1024u : 0u));
      __builtin_amdgcn_s_wait_tensorcnt(1);
    } else {
      __builtin_amdgcn_s_wait_tensorcnt(0);    // drain for the final tile
    }

    // B fragment (16-bit B 32x16): half h<8 -> K = h + 8*grp (real K 0..15),
    // h>=8 -> K = h + 8 + 8*grp >= 16 (all zero since the real K extent is 16).
    // Load UNCONDITIONALLY (address always in-bounds; elements h>=8 read a
    // redundant value) and select zero on the value — keeps the 16 ds_loads
    // in one clause with a single s_wait_dscnt instead of 16 exec-masked
    // regions.
    const float* lt = lbuf0 + ((it & 1) ? 256 : 0);        // [z][col] row-major
    half16 b;
    #pragma unroll
    for (int h = 0; h < 16; ++h) {
      float v = lt[((h & 7) + 8 * grp) * 16 + m];
      b[h] = (h < 8) ? (_Float16)v : (_Float16)0.0f;
    }

    float8 c = {};
    float8 d = __builtin_amdgcn_wmma_f32_16x16x32_f16(
        /*neg_a=*/false, a, /*neg_b=*/false, b,
        /*c_mod=*/(short)0, c, /*reuse_a=*/false, /*reuse_b=*/false);

    // D layout: VGPR v -> row M = v + 8*grp, col N = m
    #pragma unroll
    for (int v = 0; v < 8; ++v)
      out[(size_t)(v + 8 * grp) * SLICE + tile * 16 + m] = d[v];
  }
}

// ---------------- forward projector + ratio ----------------
// block = (16 z, 16 rays); one thread integrates one (z, phi, r) ray,
// then computes q = mult * data / (mult*proj + contam) * STEP in place.
// Branchless inner loop: clamp coords so the 4 bilinear loads are always
// issued, multiply the sample by the in-bounds mask (== reference's `valid`).
__global__ __launch_bounds__(256) void k_project_ratio(const float* __restrict__ vol,
                                                       const float* __restrict__ data,
                                                       const float* __restrict__ contam,
                                                       const float* __restrict__ mult,
                                                       float* __restrict__ q) {
  int z   = threadIdx.x;
  int ray = blockIdx.x * blockDim.y + threadIdx.y;
  if (ray >= SINO_Z) return;
  int phi = ray / NRAD;
  int r   = ray - phi * NRAD;

  float ang = PI_F * (float)phi / (float)NPHI;
  float sp, cp;
  __sincosf(ang, &sp, &cp);
  float rv = -200.0f + DRF * (float)r;

  const float* vz = vol + (size_t)z * SLICE;
  float acc = 0.f;
  for (int t = 0; t < NT; ++t) {
    float tv = -350.0f + 2.0f * (float)t;
    float ix = (rv * (-sp) + tv * cp - ORGF) * 0.5f;    // / VOX
    float iy = (rv * cp + tv * sp - ORGF) * 0.5f;
    float inb = (ix >= 0.f && ix <= (float)(N1 - 1) &&
                 iy >= 0.f && iy <= (float)(N2 - 1)) ? 1.f : 0.f;
    float ixc = fminf(fmaxf(ix, 0.f), (float)(N1 - 1));
    float iyc = fminf(fmaxf(iy, 0.f), (float)(N2 - 1));
    float i0f = floorf(ixc), j0f = floorf(iyc);
    float fi = ixc - i0f, fj = iyc - j0f;
    int i0 = (int)i0f, j0 = (int)j0f;
    int i1 = (i0 + 1 > N1 - 1) ? (N1 - 1) : (i0 + 1);
    int j1 = (j0 + 1 > N2 - 1) ? (N2 - 1) : (j0 + 1);
    const float* r0 = vz + i0 * N2;
    const float* r1 = vz + i1 * N2;
    float s = (1.f - fi) * (1.f - fj) * r0[j0] + fi * (1.f - fj) * r1[j0]
            + (1.f - fi) * fj * r0[j1] + fi * fj * r1[j1];
    acc += inb * s;
  }
  size_t sidx = (size_t)z * SINO_Z + (size_t)phi * NRAD + r;
  float mu = mult[sidx];
  float e  = mu * (acc * STEPF) + contam[sidx];
  q[sidx]  = mu * (data[sidx] / e) * STEPF;
}

// ---------------- exact-adjoint backprojector (gather, no atomics) ----------------
// (r,t)<->(x1,x2) is an orthonormal rotation, so the samples touching a voxel lie
// within |dr|,|dt| <= sqrt(2)*VOX of (r*,t*) = rotated voxel position.
// Branchless inner loop: tent weights clamped at 0 x in-bounds mask.
__global__ __launch_bounds__(256) void k_backproject(const float* __restrict__ q,
                                                     float* __restrict__ out) {
  __shared__ float cs[NPHI], sn[NPHI];
  int tid = threadIdx.y * 16 + threadIdx.x;
  if (tid < NPHI) {
    float a = PI_F * (float)tid / (float)NPHI;
    __sincosf(a, &sn[tid], &cs[tid]);
  }
  __syncthreads();

  int z   = threadIdx.x;
  int vox = blockIdx.x * blockDim.y + threadIdx.y;
  if (vox >= SLICE) return;
  int i1 = vox / N2, i2 = vox - i1 * N2;
  float p1 = ORGF + VOXF * (float)i1;
  float p2 = ORGF + VOXF * (float)i2;

  const float* qz = q + (size_t)z * SINO_Z;
  float acc = 0.f;
  for (int phi = 0; phi < NPHI; ++phi) {
    float cp = cs[phi], sp = sn[phi];
    float rs = -p1 * sp + p2 * cp;                      // r*
    float ts =  p1 * cp + p2 * sp;                      // t*
    int k0 = (int)ceilf((rs - 2.9f + 200.0f) / DRF);
    int k1 = (int)floorf((rs + 2.9f + 200.0f) / DRF);
    k0 = iclamp(k0, 0, NRAD - 1);  k1 = iclamp(k1, 0, NRAD - 1);
    int l0 = (int)ceilf((ts - 2.9f + 350.0f) * 0.5f);
    int l1 = (int)floorf((ts + 2.9f + 350.0f) * 0.5f);
    l0 = iclamp(l0, 0, NT - 1);    l1 = iclamp(l1, 0, NT - 1);
    const float* qrow = qz + phi * NRAD;
    for (int k = k0; k <= k1; ++k) {
      float rk = -200.0f + DRF * (float)k;
      float qv = qrow[k];
      for (int l = l0; l <= l1; ++l) {
        float tl = -350.0f + 2.0f * (float)l;
        float ix = (-rk * sp + tl * cp - ORGF) * 0.5f;
        float iy = ( rk * cp + tl * sp - ORGF) * 0.5f;
        float inb = (ix >= 0.f && ix <= (float)(N1 - 1) &&
                     iy >= 0.f && iy <= (float)(N2 - 1)) ? 1.f : 0.f;
        float w1 = fmaxf(0.f, 1.f - fabsf(ix - (float)i1));
        float w2 = fmaxf(0.f, 1.f - fabsf(iy - (float)i2));
        acc += inb * w1 * w2 * qv;
      }
    }
  }
  out[(size_t)z * SLICE + vox] = acc;
}

// ---------------- adjoint axis-1 / axis-2 smoothing (+ final elementwise) ----------------
__global__ __launch_bounds__(256) void k_adj_smooth1(const float* __restrict__ in,
                                                     float* __restrict__ out) {
  int idx = blockIdx.x * blockDim.x + threadIdx.x;
  if (idx >= NVOX) return;
  float g[KW]; make_gw(g);
  int i = (idx / N2) % N1;
  const float* col = in + (idx - i * N2);
  out[idx] = adj_axis(col, N2, i, g, N1);
}

__global__ __launch_bounds__(256) void k_adj_smooth2_final(const float* __restrict__ in,
                                                           const float* __restrict__ x,
                                                           const float* __restrict__ sens,
                                                           float* __restrict__ out) {
  int idx = blockIdx.x * blockDim.x + threadIdx.x;
  if (idx >= NVOX) return;
  float g[KW]; make_gw(g);
  int j = idx % N2;
  const float* row = in + (idx - j);
  float adj = adj_axis(row, 1, j, g, N2);
  out[idx] = x[idx] * adj / sens[idx];
}

// ---------------- host-side launcher ----------------
extern "C" void kernel_launch(void* const* d_in, const int* in_sizes, int n_in,
                              void* d_out, int out_size, void* d_ws, size_t ws_size,
                              hipStream_t stream) {
  const float* x      = (const float*)d_in[0];
  const float* data   = (const float*)d_in[1];
  const float* contam = (const float*)d_in[2];
  const float* mult   = (const float*)d_in[3];
  const float* sens   = (const float*)d_in[4];
  float* out = (float*)d_out;

  float* A = (float*)d_ws;        // volume scratch A
  float* B = A + NVOX;            // volume scratch B
  float* Q = B + NVOX;            // premultiplied ratio sinogram

  const int VB = (NVOX + 255) / 256;      // 2500 elementwise blocks
  dim3 blk2d(16, 16);

  const int zWaves  = (NTILES + TPW - 1) / TPW;   // 625 waves
  const int zBlocks = (zWaves + 7) / 8;           // 8 waves / block

  k_smooth2<<<VB, 256, 0, stream>>>(x, A);                                   // axis 2
  k_smooth1<<<VB, 256, 0, stream>>>(A, B);                                   // axis 1
  k_zfilt_wmma<<<zBlocks, 256, 0, stream>>>(B, A, 0);                        // axis 0 (WMMA)
  k_project_ratio<<<(SINO_Z + 15) / 16, blk2d, 0, stream>>>(A, data, contam, mult, Q);
  k_backproject<<<SLICE / 16, blk2d, 0, stream>>>(Q, B);                     // project^T
  k_zfilt_wmma<<<zBlocks, 256, 0, stream>>>(B, A, 1);                        // axis 0^T (WMMA)
  k_adj_smooth1<<<VB, 256, 0, stream>>>(A, B);                               // axis 1^T
  k_adj_smooth2_final<<<VB, 256, 0, stream>>>(B, x, sens, out);              // axis 2^T + MLEM
}
